// MultimodalGraphSAGE_AttentionFusion_44169443672519
// MI455X (gfx1250) — matrix operand order repaired and checked by reading, not compiled, not run on hardware
//
#include <hip/hip_runtime.h>
#include <hip/hip_bf16.h>
#include <math.h>

#define HD 128

typedef __attribute__((ext_vector_type(16))) __bf16 bf16x16;
typedef __attribute__((ext_vector_type(8)))  float  f32x8;

#define ACT_NONE 0
#define ACT_RELU 1
#define ACT_ELU  2
#define ACT_GELU 3

__device__ __forceinline__ float apply_act(float v, int act) {
  if (act == ACT_RELU) return fmaxf(v, 0.f);
  if (act == ACT_ELU)  return v > 0.f ? v : (expf(v) - 1.f);
  if (act == ACT_GELU) return 0.5f * v * (1.f + erff(v * 0.70710678118654752f));
  return v;
}

// One GEMM pass: acc[t] += (rowscale ? A*rs : A) @ W^T tile t, bf16 WMMA, f32 acc.
// A fragment loaded ONCE per K-step and reused across all NTILES WMMAs.
// Layouts per ISA 7.12.2:
//   A (16x32 bf16): lane&15 = M row; element i -> K = (i>>3)*16 + (lane>>4)*8 + (i&7)
//   B (32x16 bf16): lane&15 = N col; element i -> K = (i>>3)*16 + (lane>>4)*8 + (i&7)
template<int NTILES>
__device__ __forceinline__ void gemm_pass(f32x8* acc,
                                          const float* __restrict__ A,
                                          const float* __restrict__ W,
                                          const float* __restrict__ rowscale,
                                          long m0, int lane, int k)
{
  const int g  = lane >> 4;
  const int lm = lane & 15;
  const float* arow  = A + (m0 + lm) * (long)k;
  const float* wbase = W + (long)lm * k;
  float invc = 1.f;
  if (rowscale) invc = 1.f / fmaxf(rowscale[m0 + lm], 1.f);

  for (int kk = 0; kk < k; kk += 32) {
    bf16x16 av;
#pragma unroll
    for (int i = 0; i < 16; ++i) {
      const int ka = kk + ((i >> 3) << 4) + (g << 3) + (i & 7);
      av[i] = (__bf16)(arow[ka] * invc);
    }
#pragma unroll
    for (int t = 0; t < NTILES; ++t) {
      const float* wrow = wbase + (long)t * 16 * k;
      bf16x16 bv;
#pragma unroll
      for (int i = 0; i < 16; ++i) {
        const int kb = kk + ((i >> 3) << 4) + (g << 3) + (i & 7);
        bv[i] = (__bf16)wrow[kb];
      }
      acc[t] = __builtin_amdgcn_wmma_f32_16x16x32_bf16(false, av, false, bv,
                                                       (short)0, acc[t], false, false);
    }
  }
}

// C[n x (16*NTILES)] = act( A[n x k] @ W^T (+ A2[n x k2] @ W2^T) + bias )
// W row-major [dout x k] (Linear layout).  rowscale: per-A-row 1/max(cnt,1) fold-in.
// One wave32 computes a full 16 x (16*NTILES) slab of C.
template<int NTILES>
__global__ __launch_bounds__(32) void wmma_gemm_kernel(
    const float* __restrict__ A,  const float* __restrict__ W,
    const float* __restrict__ rowscale,
    const float* __restrict__ A2, const float* __restrict__ W2,
    const float* __restrict__ bias, float* __restrict__ C,
    int k, int k2, int act)
{
  constexpr int DOUT = 16 * NTILES;
  const int lane = threadIdx.x & 31;
  const int g    = lane >> 4;
  const int lm   = lane & 15;
  const long m0  = (long)blockIdx.x * 16;

  f32x8 acc[NTILES];
#pragma unroll
  for (int t = 0; t < NTILES; ++t) acc[t] = (f32x8){0.f,0.f,0.f,0.f,0.f,0.f,0.f,0.f};

  gemm_pass<NTILES>(acc, A, W, rowscale, m0, lane, k);
  if (A2 != nullptr) gemm_pass<NTILES>(acc, A2, W2, nullptr, m0, lane, k2);

#pragma unroll
  for (int t = 0; t < NTILES; ++t) {
    const int col = t * 16 + lm;
    const float bb = bias ? bias[col] : 0.f;
#pragma unroll
    for (int r = 0; r < 8; ++r) {
      const long row = m0 + r + (g << 3);
      C[row * (long)DOUT + col] = apply_act(acc[t][r] + bb, act);
    }
  }
}

// per-column mean / rstd over nrows (batchnorm stats), one block per column
__global__ void colstats_kernel(const float* __restrict__ x, int nrows,
                                float* __restrict__ mean, float* __restrict__ rstd)
{
  __shared__ float ss[256];
  __shared__ float sq[256];
  const int c = blockIdx.x;   // 0..HD-1
  float s = 0.f, q = 0.f;
  for (int i = threadIdx.x; i < nrows; i += blockDim.x) {
    float v = x[(long)i * HD + c];
    s += v; q += v * v;
  }
  ss[threadIdx.x] = s; sq[threadIdx.x] = q;
  __syncthreads();
  for (int off = blockDim.x >> 1; off > 0; off >>= 1) {
    if ((int)threadIdx.x < off) {
      ss[threadIdx.x] += ss[threadIdx.x + off];
      sq[threadIdx.x] += sq[threadIdx.x + off];
    }
    __syncthreads();
  }
  if (threadIdx.x == 0) {
    float m   = ss[0] / (float)nrows;
    float var = sq[0] / (float)nrows - m * m;
    mean[c] = m;
    rstd[c] = rsqrtf(var + 1e-5f);
  }
}

__global__ void bn_relu_kernel(const float* __restrict__ x,
                               const float* __restrict__ mean,
                               const float* __restrict__ rstd,
                               const float* __restrict__ gw,
                               const float* __restrict__ bw,
                               float* __restrict__ y, long total)
{
  long tid = (long)blockIdx.x * blockDim.x + threadIdx.x;
  if (tid >= total) return;
  int c = (int)(tid & (HD - 1));
  float v = gw[c] * (x[tid] - mean[c]) * rstd[c] + bw[c];
  y[tid] = fmaxf(v, 0.f);
}

__global__ void zero_kernel(float* __restrict__ p, long total) {
  long tid = (long)blockIdx.x * blockDim.x + threadIdx.x;
  if (tid < total) p[tid] = 0.f;
}

// edges: [2 x E] int32, row0 = src, row1 = dst.  agg[dst] += x[src]; cnt[dst]+=1
__global__ void scatter_add_kernel(const float* __restrict__ x,
                                   const int* __restrict__ edges,
                                   float* __restrict__ agg,
                                   float* __restrict__ cnt, int E)
{
  long tid = (long)blockIdx.x * blockDim.x + threadIdx.x;
  long total = (long)E * HD;
  if (tid >= total) return;
  int e = (int)(tid >> 7);
  int f = (int)(tid & (HD - 1));
  int src = edges[e];
  int dst = edges[E + e];
  atomicAdd(&agg[(long)dst * HD + f], x[(long)src * HD + f]);
  if (f == 0) atomicAdd(&cnt[dst], 1.0f);
}

// one wave32 per node: attention scores (dot-64 via shuffle), softmax over 3,
// fused feature, classifier dot-128, sigmoid. Writes preds[n] then alpha[n x 3].
__global__ __launch_bounds__(32) void fuse_kernel(
    const float* __restrict__ xall, const float* __restrict__ hid,
    const float* __restrict__ w2a, const float* __restrict__ b2a,
    const float* __restrict__ w2b, const float* __restrict__ b2b,
    const float* __restrict__ w2c, const float* __restrict__ b2c,
    const float* __restrict__ clsW, const float* __restrict__ clsb,
    float* __restrict__ out, int n)
{
  const int node = blockIdx.x;
  const int lane = threadIdx.x & 31;
  float sc[3];
#pragma unroll
  for (int m = 0; m < 3; ++m) {
    const float* h  = hid + ((long)m * n + node) * 64;
    const float* w2 = (m == 0) ? w2a : (m == 1) ? w2b : w2c;
    const float* b2 = (m == 0) ? b2a : (m == 1) ? b2b : b2c;
    float p = h[lane] * w2[lane] + h[lane + 32] * w2[lane + 32];
    for (int off = 16; off > 0; off >>= 1) p += __shfl_xor(p, off, 32);
    sc[m] = p + b2[0];
  }
  float mx = fmaxf(sc[0], fmaxf(sc[1], sc[2]));
  float e0 = expf(sc[0] - mx), e1 = expf(sc[1] - mx), e2 = expf(sc[2] - mx);
  float inv = 1.f / (e0 + e1 + e2);
  float a0 = e0 * inv, a1 = e1 * inv, a2 = e2 * inv;

  float lp = 0.f;
#pragma unroll
  for (int t = 0; t < 4; ++t) {
    int c = lane + t * 32;
    float f = a0 * xall[(long)node * HD + c]
            + a1 * xall[((long)n + node) * HD + c]
            + a2 * xall[((long)2 * n + node) * HD + c];
    lp += f * clsW[c];
  }
  for (int off = 16; off > 0; off >>= 1) lp += __shfl_xor(lp, off, 32);
  if (lane == 0) {
    float logit = lp + clsb[0];
    out[node] = 1.f / (1.f + expf(-logit));
    out[n + (long)node * 3 + 0] = a0;
    out[n + (long)node * 3 + 1] = a1;
    out[n + (long)node * 3 + 2] = a2;
  }
}

static inline unsigned cdivu(long a, long b) { return (unsigned)((a + b - 1) / b); }

extern "C" void kernel_launch(void* const* d_in, const int* in_sizes, int n_in,
                              void* d_out, int out_size, void* d_ws, size_t ws_size,
                              hipStream_t stream)
{
  (void)n_in; (void)out_size; (void)ws_size;
  const int n    = in_sizes[0] / 512;   // 20000
  const int n3   = 3 * n;               // 60000
  const int Emod = in_sizes[48] / 2;    // 320000
  const int Eint = in_sizes[51] / 2;    // 600000

  const float* xin[3] = { (const float*)d_in[0], (const float*)d_in[1], (const float*)d_in[2] };
  const int    din[3] = { 512, 256, 384 };

  float* B0   = (float*)d_ws;                 // h_all / x_all ping
  float* B1   = B0 + (long)n3 * HD;           // tmp / x_all pong (final)
  float* B2   = B1 + (long)n3 * HD;           // agg (neighbor sums)
  float* B3   = B2 + (long)n3 * HD;           // xs / hid
  float* cnt  = B3 + (long)n3 * HD;           // n3 floats (neighbor counts)
  float* mean = cnt + n3;                     // HD
  float* rstd = mean + HD;                    // HD

  // ---- projections + batchnorm + relu  ->  B0 (h1,h2,h3 stacked) ----
  for (int m = 0; m < 3; ++m) {
    const float* pW = (const float*)d_in[3 + 2 * m];
    const float* pb = (const float*)d_in[4 + 2 * m];
    const float* bg = (const float*)d_in[9 + 2 * m];
    const float* bb = (const float*)d_in[10 + 2 * m];
    float* slab1 = B1 + (long)m * n * HD;
    float* slab0 = B0 + (long)m * n * HD;
    wmma_gemm_kernel<8><<<n / 16, 32, 0, stream>>>(
        xin[m], pW, nullptr, nullptr, nullptr, pb, slab1, din[m], 0, ACT_NONE);
    colstats_kernel<<<HD, 256, 0, stream>>>(slab1, n, mean, rstd);
    bn_relu_kernel<<<cdivu((long)n * HD, 256), 256, 0, stream>>>(
        slab1, mean, rstd, bg, bb, slab0, (long)n * HD);
  }

  // ---- per-modality SAGE conv (elu)  ->  B1 (x_all) ----
  // mean-divide is folded into the GEMM A-load via rowscale = cnt
  for (int m = 0; m < 3; ++m) {
    const float* lW = (const float*)d_in[15 + 3 * m];
    const float* lb = (const float*)d_in[16 + 3 * m];
    const float* rW = (const float*)d_in[17 + 3 * m];
    const int*   ed = (const int*)d_in[48 + m];
    const float* xs = B0 + (long)m * n * HD;
    float* agg   = B2 + (long)m * n * HD;
    float* cslab = cnt + (long)m * n;
    zero_kernel<<<cdivu((long)n * HD, 256), 256, 0, stream>>>(agg, (long)n * HD);
    zero_kernel<<<cdivu(n, 256), 256, 0, stream>>>(cslab, n);
    scatter_add_kernel<<<cdivu((long)Emod * HD, 256), 256, 0, stream>>>(xs, ed, agg, cslab, Emod);
    wmma_gemm_kernel<8><<<n / 16, 32, 0, stream>>>(
        agg, lW, cslab, xs, rW, lb, B1 + (long)m * n * HD, HD, HD, ACT_ELU);
  }

  // ---- two inter SAGE convs (relu-proj, mean agg folded into GEMM, gelu) ----
  const int* ei = (const int*)d_in[51];
  for (int it = 0; it < 2; ++it) {
    const float* pW = (const float*)d_in[24 + 5 * it];
    const float* pb = (const float*)d_in[25 + 5 * it];
    const float* lW = (const float*)d_in[26 + 5 * it];
    const float* lb = (const float*)d_in[27 + 5 * it];
    const float* rW = (const float*)d_in[28 + 5 * it];
    const float* src = (it == 0) ? B1 : B0;
    float*       dst = (it == 0) ? B0 : B1;   // final x_all lands in B1
    wmma_gemm_kernel<8><<<n3 / 16, 32, 0, stream>>>(
        src, pW, nullptr, nullptr, nullptr, pb, B3, HD, 0, ACT_RELU);
    zero_kernel<<<cdivu((long)n3 * HD, 256), 256, 0, stream>>>(B2, (long)n3 * HD);
    zero_kernel<<<cdivu(n3, 256), 256, 0, stream>>>(cnt, n3);
    scatter_add_kernel<<<cdivu((long)Eint * HD, 256), 256, 0, stream>>>(B3, ei, B2, cnt, Eint);
    wmma_gemm_kernel<8><<<n3 / 16, 32, 0, stream>>>(
        B2, lW, cnt, src, rW, lb, dst, HD, HD, ACT_GELU);
  }

  // ---- attention hidden layers (dout=64, relu) into B3 (xs buffer now free) ----
  for (int m = 0; m < 3; ++m) {
    const float* w1 = (const float*)d_in[34 + 4 * m];
    const float* b1 = (const float*)d_in[35 + 4 * m];
    wmma_gemm_kernel<4><<<n / 16, 32, 0, stream>>>(
        B1 + (long)m * n * HD, w1, nullptr, nullptr, nullptr, b1,
        B3 + (long)m * n * 64, HD, 0, ACT_RELU);
  }

  // ---- softmax fusion + classifier + sigmoid -> d_out (preds, alpha) ----
  fuse_kernel<<<n, 32, 0, stream>>>(
      B1, B3,
      (const float*)d_in[36], (const float*)d_in[37],
      (const float*)d_in[40], (const float*)d_in[41],
      (const float*)d_in[44], (const float*)d_in[45],
      (const float*)d_in[46], (const float*)d_in[47],
      (float*)d_out, n);
}
